// MultiheadAttention_39393440039034
// MI455X (gfx1250) — matrix operand order repaired and checked by reading, not compiled, not run on hardware
//
#include <hip/hip_runtime.h>
#include <hip/hip_bf16.h>

typedef __attribute__((ext_vector_type(16))) _Float16 v16h;
typedef __attribute__((ext_vector_type(8)))  _Float16 v8h;
typedef __attribute__((ext_vector_type(8)))  float    v8f;
typedef __attribute__((ext_vector_type(4)))  int      v4i;

#define B_   4
#define SQ_  2048
#define SKV_ 2048
#define ED_  1024
#define NH_  16
#define HD_  64

#define GAS __attribute__((address_space(1)))
#define LAS __attribute__((address_space(3)))

#if defined(__gfx1250__) && \
    __has_builtin(__builtin_amdgcn_global_load_async_to_lds_b128) && \
    __has_builtin(__builtin_amdgcn_s_wait_asynccnt)
#define USE_ASYNC 1
#else
#define USE_ASYNC 0
#endif

__device__ __forceinline__ v8f wmma_f16(v16h a, v16h b, v8f c) {
  // D = A(16x32 f16) * B(32x16 f16) + C(16x16 f32)
  return __builtin_amdgcn_wmma_f32_16x16x32_f16(false, a, false, b, (short)0, c,
                                                false, false);
}

// A fragment: 16(M) x 32(K) f16 from row-major storage with leading dim ld (halves).
// ISA layout: lane m<16 holds row m, K {k0..k0+7, k0+16..k0+23}, k0 = 0 (lanes<16) or 8.
__device__ __forceinline__ v16h load_fragA(const _Float16* __restrict__ p0, int ld, int lane) {
  const int r  = lane & 15;
  const int k0 = (lane & 16) ? 8 : 0;
  const _Float16* p = p0 + (size_t)r * ld + k0;
  v8h lo = *reinterpret_cast<const v8h*>(p);
  v8h hi = *reinterpret_cast<const v8h*>(p + 16);
  v16h f;
#pragma unroll
  for (int i = 0; i < 8; ++i) { f[i] = lo[i]; f[i + 8] = hi[i]; }
  return f;
}

// B fragment: 32(K) x 16(N) f16, loaded from Bt storage [N x K] row-major (column-contig).
// ISA layout: lane n<16 holds column n, K {k0..k0+7, k0+16..k0+23}, k0 = 0 (lanes<16) or 8.
__device__ __forceinline__ v16h load_fragB(const _Float16* __restrict__ p0, int ld, int lane) {
  const int n  = lane & 15;
  const int k0 = (lane & 16) ? 8 : 0;
  const _Float16* p = p0 + (size_t)n * ld + k0;
  v8h lo = *reinterpret_cast<const v8h*>(p);
  v8h hi = *reinterpret_cast<const v8h*>(p + 16);
  v16h f;
#pragma unroll
  for (int i = 0; i < 8; ++i) { f[i] = lo[i]; f[i + 8] = hi[i]; }
  return f;
}

__global__ void cvt_f32_f16(const float* __restrict__ in, _Float16* __restrict__ out, int n) {
  int i = blockIdx.x * blockDim.x + threadIdx.x;
  const int stride = gridDim.x * blockDim.x;
  for (; i < n; i += stride) out[i] = (_Float16)in[i];
}

// in: [rows x cols] f32 row-major -> out: [cols x rows] f16 row-major (transpose)
__global__ void cvt_transpose_f32_f16(const float* __restrict__ in, _Float16* __restrict__ out,
                                      int rows, int cols) {
  int i = blockIdx.x * blockDim.x + threadIdx.x;
  const int n = rows * cols;
  const int stride = gridDim.x * blockDim.x;
  for (; i < n; i += stride) {
    const int r = i / cols;
    const int c = i - r * cols;
    out[(size_t)c * rows + r] = (_Float16)in[i];
  }
}

// kv = c @ W_kv + b_kv, scattered into K[b,h,skv,hd] (f16) and Vt[b,h,hd,skv] (f16)
__global__ void __launch_bounds__(256)
kv_proj_kernel(const _Float16* __restrict__ A,   // [B*SKV, ED] f16  (context c)
               const _Float16* __restrict__ Wt,  // [2*ED, ED] f16   (W_kv^T)
               const float*    __restrict__ bias,// [2*ED]
               _Float16* __restrict__ Kh,        // [B,NH,SKV,HD]
               _Float16* __restrict__ Vt)        // [B,NH,HD,SKV]
{
  const int lane = threadIdx.x & 31;
  const int wave = threadIdx.x >> 5;
  const int m0 = blockIdx.y * 128 + wave * 16;
  const int n0 = blockIdx.x * 64;
  v8f acc[4] = {{}, {}, {}, {}};
  const _Float16* arow = A + (size_t)m0 * ED_;
#pragma unroll 4
  for (int k = 0; k < ED_; k += 32) {
    v16h a = load_fragA(arow + k, ED_, lane);
#pragma unroll
    for (int t = 0; t < 4; ++t) {
      v16h b = load_fragB(Wt + (size_t)(n0 + t * 16) * ED_ + k, ED_, lane);
      acc[t] = wmma_f16(a, b, acc[t]);
    }
  }
  const int colL  = lane & 15;
  const int rbase = (lane & 16) ? 8 : 0;
#pragma unroll
  for (int t = 0; t < 4; ++t) {
    const int col = n0 + t * 16 + colL;
    const int h = col >> 7;        // / (2*HD)
    const int r = col & 127;       // % (2*HD)
    const float bia = bias[col];
#pragma unroll
    for (int i = 0; i < 8; ++i) {
      const int row = m0 + rbase + i;
      const int bb  = row >> 11;         // / SKV_
      const int skv = row & (SKV_ - 1);
      const float val = acc[t][i] + bia;
      if (r < HD_) {
        Kh[((size_t)(bb * NH_ + h) * SKV_ + skv) * HD_ + r] = (_Float16)val;
      } else {
        Vt[((size_t)(bb * NH_ + h) * HD_ + (r - HD_)) * SKV_ + skv] = (_Float16)val;
      }
    }
  }
}

// Cooperative staging of one 64-row K chunk and one 64-col V chunk into LDS.
// kg: K chunk base, rows kv (ld HD_); vg: Vt chunk base, rows d (ld SKV_).
// LDS tiles are [64][LDP] halves; 2 x b128 per thread per tile.
template <int LDP>
__device__ __forceinline__ void stage_chunk(const _Float16* kg, const _Float16* vg,
                                            _Float16* kl, _Float16* vl, int tid) {
#pragma unroll
  for (int t = 0; t < 2; ++t) {
    const int e = t * 256 + tid;   // 0..511
    const int r = e >> 3;          // row 0..63
    const int g = (e & 7) * 8;     // column group (halves)
#if USE_ASYNC
    __builtin_amdgcn_global_load_async_to_lds_b128(
        (GAS v4i*)(kg + (size_t)r * HD_ + g), (LAS v4i*)(kl + r * LDP + g), 0, 0);
    __builtin_amdgcn_global_load_async_to_lds_b128(
        (GAS v4i*)(vg + (size_t)r * SKV_ + g), (LAS v4i*)(vl + r * LDP + g), 0, 0);
#else
    *reinterpret_cast<v8h*>(kl + r * LDP + g) =
        *reinterpret_cast<const v8h*>(kg + (size_t)r * HD_ + g);
    *reinterpret_cast<v8h*>(vl + r * LDP + g) =
        *reinterpret_cast<const v8h*>(vg + (size_t)r * SKV_ + g);
#endif
  }
}

// Flash attention: block = 8 waves = 128 query rows of one (b,h).
// K/V chunks (64 KV rows) staged once per block into LDS (async, double buffered).
__global__ void __launch_bounds__(256)
attn_kernel(const _Float16* __restrict__ Qh,  // [B,SQ,ED] f16
            const _Float16* __restrict__ Kh,  // [B,NH,SKV,HD] f16
            const _Float16* __restrict__ Vt,  // [B,NH,HD,SKV] f16
            _Float16* __restrict__ Oh)        // [B,SQ,ED] f16
{
  constexpr int LDP = 72;  // padded LDS pitch (halves): 144B rows, 16B aligned
  __shared__ __align__(16) _Float16 kbuf[2][64 * LDP];
  __shared__ __align__(16) _Float16 vbuf[2][64 * LDP];
  __shared__ __align__(16) _Float16 pbuf[8][16 * LDP];
  const int tid  = threadIdx.x;
  const int lane = tid & 31;
  const int wave = tid >> 5;
  const int bh = blockIdx.y;
  const int b  = bh >> 4;
  const int h  = bh & 15;
  const int m0 = blockIdx.x * 128 + wave * 16;

  const _Float16* qp = Qh + (size_t)(b * SQ_ + m0) * ED_ + h * HD_;
  const v16h aq0 = load_fragA(qp, ED_, lane);       // d 0..31
  const v16h aq1 = load_fragA(qp + 32, ED_, lane);  // d 32..63

  const _Float16* kbase = Kh + (size_t)bh * SKV_ * HD_;
  const _Float16* vbase = Vt + (size_t)bh * HD_ * SKV_;

  float mrow[8], lrow[8], corr[8];
#pragma unroll
  for (int i = 0; i < 8; ++i) { mrow[i] = -1e30f; lrow[i] = 0.0f; }
  v8f o[4] = {{}, {}, {}, {}};
  const float scale = 0.125f;  // 1/sqrt(64)
  const int colL = lane & 15;
  const int rsel = (lane & 16) ? 8 : 0;
  _Float16* pw = &pbuf[wave][0];

  stage_chunk<LDP>(kbase, vbase, &kbuf[0][0], &vbuf[0][0], tid);
  int cur = 0;
  for (int j = 0; j < SKV_; j += 64) {
#if USE_ASYNC
    __builtin_amdgcn_s_wait_asynccnt(0);  // my DMA for buf[cur] done
#endif
    __syncthreads();                      // everyone's DMA done; prev readers done
    if (j + 64 < SKV_) {
      stage_chunk<LDP>(kbase + (size_t)(j + 64) * HD_, vbase + (j + 64),
                       &kbuf[cur ^ 1][0], &vbuf[cur ^ 1][0], tid);
    }
    const _Float16* kl = &kbuf[cur][0];
    const _Float16* vl = &vbuf[cur][0];

    // S = Q * K^T for 4 column tiles (64 KV), k-split over d = 2 x 32
    v8f s[4];
#pragma unroll
    for (int n = 0; n < 4; ++n) {
      v16h b0 = load_fragB(kl + n * 16 * LDP, LDP, lane);
      v16h b1 = load_fragB(kl + n * 16 * LDP + 32, LDP, lane);
      const v8f z = {};
      s[n] = wmma_f16(aq0, b0, z);
      s[n] = wmma_f16(aq1, b1, s[n]);
    }

    // online softmax: row = rsel + i lives across the lane's 16-lane half
#pragma unroll
    for (int i = 0; i < 8; ++i) {
      const float a0 = s[0][i] * scale;
      const float a1 = s[1][i] * scale;
      const float a2 = s[2][i] * scale;
      const float a3 = s[3][i] * scale;
      float v = fmaxf(fmaxf(a0, a1), fmaxf(a2, a3));
      v = fmaxf(v, __shfl_xor(v, 1, 32));
      v = fmaxf(v, __shfl_xor(v, 2, 32));
      v = fmaxf(v, __shfl_xor(v, 4, 32));
      v = fmaxf(v, __shfl_xor(v, 8, 32));
      const float mn = fmaxf(mrow[i], v);
      const float p0 = __expf(a0 - mn);
      const float p1 = __expf(a1 - mn);
      const float p2 = __expf(a2 - mn);
      const float p3 = __expf(a3 - mn);
      corr[i] = __expf(mrow[i] - mn);
      mrow[i] = mn;
      float rs = (p0 + p1) + (p2 + p3);
      rs += __shfl_xor(rs, 1, 32);
      rs += __shfl_xor(rs, 2, 32);
      rs += __shfl_xor(rs, 4, 32);
      rs += __shfl_xor(rs, 8, 32);
      lrow[i] = lrow[i] * corr[i] + rs;
      // D-layout -> row-major LDS [16][LDP]; per-wave region, LDS in-order per wave
      _Float16* pr = pw + (rsel + i) * LDP + colL;
      pr[0]  = (_Float16)p0;
      pr[16] = (_Float16)p1;
      pr[32] = (_Float16)p2;
      pr[48] = (_Float16)p3;
    }
    const v16h ap0 = load_fragA(pw, LDP, lane);       // kv 0..31 of chunk
    const v16h ap1 = load_fragA(pw + 32, LDP, lane);  // kv 32..63 of chunk
#pragma unroll
    for (int n = 0; n < 4; ++n) {
#pragma unroll
      for (int i = 0; i < 8; ++i) o[n][i] *= corr[i];
    }
#pragma unroll
    for (int n = 0; n < 4; ++n) {
      v16h bv0 = load_fragB(vl + n * 16 * LDP, LDP, lane);
      v16h bv1 = load_fragB(vl + n * 16 * LDP + 32, LDP, lane);
      o[n] = wmma_f16(ap0, bv0, o[n]);
      o[n] = wmma_f16(ap1, bv1, o[n]);
    }
    cur ^= 1;
  }

  _Float16* op = Oh + (size_t)(b * SQ_ + m0) * ED_ + h * HD_;
#pragma unroll
  for (int n = 0; n < 4; ++n) {
#pragma unroll
    for (int i = 0; i < 8; ++i) {
      const float inv = 1.0f / lrow[i];
      op[(size_t)(rsel + i) * ED_ + n * 16 + colL] = (_Float16)(o[n][i] * inv);
    }
  }
}

// out = vals @ W_o + b_o  (f32 output)
__global__ void __launch_bounds__(256)
out_proj_kernel(const _Float16* __restrict__ A,   // [B*SQ, ED] f16
                const _Float16* __restrict__ Wt,  // [ED, ED] f16 (W_o^T)
                const float*    __restrict__ bias,// [ED]
                float* __restrict__ C)            // [B*SQ, ED] f32
{
  const int lane = threadIdx.x & 31;
  const int wave = threadIdx.x >> 5;
  const int m0 = blockIdx.y * 128 + wave * 16;
  const int n0 = blockIdx.x * 64;
  v8f acc[4] = {{}, {}, {}, {}};
  const _Float16* arow = A + (size_t)m0 * ED_;
#pragma unroll 4
  for (int k = 0; k < ED_; k += 32) {
    v16h a = load_fragA(arow + k, ED_, lane);
#pragma unroll
    for (int t = 0; t < 4; ++t) {
      v16h b = load_fragB(Wt + (size_t)(n0 + t * 16) * ED_ + k, ED_, lane);
      acc[t] = wmma_f16(a, b, acc[t]);
    }
  }
  const int colL  = lane & 15;
  const int rbase = (lane & 16) ? 8 : 0;
#pragma unroll
  for (int t = 0; t < 4; ++t) {
    const int col = n0 + t * 16 + colL;
    const float bia = bias[col];
#pragma unroll
    for (int i = 0; i < 8; ++i) {
      C[(size_t)(m0 + rbase + i) * ED_ + col] = acc[t][i] + bia;
    }
  }
}

extern "C" void kernel_launch(void* const* d_in, const int* in_sizes, int n_in,
                              void* d_out, int out_size, void* d_ws, size_t ws_size,
                              hipStream_t stream) {
  (void)in_sizes; (void)n_in; (void)out_size; (void)ws_size;
  const float* q   = (const float*)d_in[0];
  const float* c   = (const float*)d_in[1];
  const float* Wkv = (const float*)d_in[2];
  const float* bkv = (const float*)d_in[3];
  const float* Wo  = (const float*)d_in[4];
  const float* bo  = (const float*)d_in[5];
  float* out = (float*)d_out;

  char* p = (char*)d_ws;
  auto carve = [&](size_t elems) {
    _Float16* r = (_Float16*)p;
    p += elems * sizeof(_Float16);
    return r;
  };
  const size_t NQ = (size_t)B_ * SQ_ * ED_;   // 8M elems
  const size_t NC = (size_t)B_ * SKV_ * ED_;  // 8M elems
  _Float16* qh   = carve(NQ);                     // q in f16
  _Float16* ch   = carve(NC);                     // c in f16
  _Float16* WtKV = carve((size_t)2 * ED_ * ED_);  // W_kv^T [2048][1024]
  _Float16* WtO  = carve((size_t)ED_ * ED_);      // W_o^T  [1024][1024]
  _Float16* Kh   = carve(NC);                     // K  [B,NH,SKV,HD]
  _Float16* Vt   = carve(NC);                     // V^T[B,NH,HD,SKV]
  _Float16* Ah   = carve(NQ);                     // attention output f16

  cvt_f32_f16<<<dim3(2048), dim3(256), 0, stream>>>(q, qh, (int)NQ);
  cvt_f32_f16<<<dim3(2048), dim3(256), 0, stream>>>(c, ch, (int)NC);
  cvt_transpose_f32_f16<<<dim3(2048), dim3(256), 0, stream>>>(Wkv, WtKV, ED_, 2 * ED_);
  cvt_transpose_f32_f16<<<dim3(1024), dim3(256), 0, stream>>>(Wo, WtO, ED_, ED_);

  kv_proj_kernel<<<dim3((2 * ED_) / 64, (B_ * SKV_) / 128), dim3(256), 0, stream>>>(
      ch, WtKV, bkv, Kh, Vt);
  attn_kernel<<<dim3(SQ_ / 128, B_ * NH_), dim3(256), 0, stream>>>(qh, Kh, Vt, Ah);
  out_proj_kernel<<<dim3(ED_ / 64, (B_ * SQ_) / 128), dim3(256), 0, stream>>>(
      Ah, WtO, bo, out);
}
